// Representation_84447646974226
// MI455X (gfx1250) — matrix operand order repaired and checked by reading, not compiled, not run on hardware
//
#include <hip/hip_runtime.h>
#include <cstdint>

// ---------------------------------------------------------------------------
// GNN forward (SAGE + DotGAT residual blocks) for gfx1250 (MI455X).
// Dense GEMMs: operands pre-padded + pre-converted to f16 in workspace, tiles
// staged to LDS with global_load_async_to_lds_b128 (ASYNCcnt, double-buffered),
// compute with v_wmma_f32_16x16x32_f16 (f32 accumulate).
// Graph ops: global_atomic_add_f32 / atomicMax(uint key) segment ops.
// ---------------------------------------------------------------------------

#define HF    512
#define NHEAD 10
#define DPH   52
#define FEAT  (NHEAD * DPH)   // 520
#define FIN   74
#define CHUNK 2048            // FFN row chunk (bounds workspace)
#define MP    20032           // 20000 rounded up to 64

typedef _Float16 v16h __attribute__((ext_vector_type(16)));
typedef _Float16 h8   __attribute__((ext_vector_type(8)));
typedef float    v8f  __attribute__((ext_vector_type(8)));

__device__ __forceinline__ float elu1(float v) {
    return v > 0.f ? v : (__expf(v) - 1.f);
}

// order-preserving float <-> uint key (for atomicMax-based segment max)
__device__ __forceinline__ unsigned fkey(float x) {
    unsigned b = __float_as_uint(x);
    return (b & 0x80000000u) ? ~b : (b | 0x80000000u);
}
__device__ __forceinline__ float fkeyinv(unsigned k) {
    unsigned b = (k & 0x80000000u) ? (k & 0x7FFFFFFFu) : ~k;
    return __uint_as_float(b);
}

// async 16B global -> LDS copy (CDNA5, ASYNCcnt-tracked)
__device__ __forceinline__ void async_b128(unsigned lds_byte, int gl_byte_off,
                                           const _Float16* base) {
    asm volatile("global_load_async_to_lds_b128 %0, %1, %2"
                 :: "v"(lds_byte), "v"(gl_byte_off), "s"(base)
                 : "memory");
}

// ---------------------------------------------------------------------------
// WMMA GEMM on pre-padded f16 operands:
//   A [Mpad x Kp] f16 (Kp % 32 == 0), B [Kp x Npad] f16 (Npad % 64 == 0, zero
//   padded), C [M x N] f32. No guards in the K loop; epilogue guards real M/N.
//   epilogue: v = acc (+bias[n]); if actElu: v = elu(v); v += res1?; v += res2?
// Block: 256 thr = 8 waves; block tile 64x64; wave tile 16x32 (2 WMMA/k-step).
// Double-buffered LDS fed by global_load_async_to_lds_b128 (1 per thread/tile).
// ---------------------------------------------------------------------------
__global__ __launch_bounds__(256) void k_gemm16(
    const _Float16* __restrict__ A, int lda,
    const _Float16* __restrict__ B, int ldb,
    float* __restrict__ C, int ldc,
    const float* __restrict__ bias,
    const float* __restrict__ res1,
    const float* __restrict__ res2,
    int M, int N, int Kp, int actElu)
{
    __shared__ __align__(32) _Float16 As[2][64 * 40];  // stride 40 halves
    __shared__ __align__(32) _Float16 Bs[2][32 * 80];  // stride 80 halves

    const int tid     = threadIdx.x;
    const int rowBase = blockIdx.y * 64;
    const int colBase = blockIdx.x * 64;
    const int lane    = tid & 31;
    const int w       = tid >> 5;
    const int wm      = (w & 3) * 16;
    const int wn      = (w >> 2) * 32;
    const int arow    = tid >> 2, acb = (tid & 3) * 8;  // A: 64x32 halves
    const int brow    = tid >> 3, bcb = (tid & 7) * 8;  // B: 32x64 halves

    // LDS byte addresses for this thread's 16B staging slot
    const unsigned ldsA = (unsigned)(uintptr_t)(&As[0][0]) + (unsigned)((arow * 40 + acb) * 2);
    const unsigned ldsB = (unsigned)(uintptr_t)(&Bs[0][0]) + (unsigned)((brow * 80 + bcb) * 2);
    const unsigned stA  = (unsigned)(64 * 40 * 2);  // stage stride bytes
    const unsigned stB  = (unsigned)(32 * 80 * 2);
    // global byte offsets (k0-dependent part added per stage)
    const int aG = (int)(((size_t)(rowBase + arow) * (size_t)lda + (size_t)acb) * 2);
    const int bG = (int)(((size_t)brow * (size_t)ldb + (size_t)(colBase + bcb)) * 2);
    const int aStep = 2 * 32;          // bytes per K-stage along A row
    const int bStep = 2 * 32 * ldb;    // bytes per K-stage down B

    v8f c0 = {};
    v8f c1 = {};

    // prologue: stage 0 into buffer 0
    async_b128(ldsA, aG, A);
    async_b128(ldsB, bG, B);

    int buf = 0;
    for (int k0 = 0; k0 < Kp; k0 += 32) {
        const bool hasNext = (k0 + 32) < Kp;
        if (hasNext) {
            const int k1 = k0 + 32;
            async_b128(ldsA + (unsigned)(buf ^ 1) * stA, aG + (k1 >> 5) * aStep, A);
            async_b128(ldsB + (unsigned)(buf ^ 1) * stB, bG + (k1 >> 5) * bStep, B);
            asm volatile("s_wait_asynccnt 0x2" ::: "memory");  // stage k0 done
        } else {
            asm volatile("s_wait_asynccnt 0x0" ::: "memory");
        }
        __syncthreads();

        // fragments (layouts per ISA 7.12.2) + WMMA
        {
            const _Float16* as = As[buf];
            const _Float16* bs = Bs[buf];
            union { v16h v; h8 h[2]; } a;
            const int m    = wm + (lane & 15);
            const int koff = (lane >> 4) * 8;                 // lanes 16-31: +8
            a.h[0] = *(const h8*)(as + m * 40 + koff);        // K 0-7  / 8-15
            a.h[1] = *(const h8*)(as + m * 40 + 16 + koff);   // K 16-23 / 24-31
            v16h b0, b1;                                              // n=lane%16, k as A
#pragma unroll
            for (int s = 0; s < 16; ++s) {
                const int kk = (s < 8) ? (koff + s) : (16 + koff + (s - 8));
                b0[s] = bs[kk * 80 + wn + (lane & 15)];
                b1[s] = bs[kk * 80 + wn + 16 + (lane & 15)];
            }
            c0 = __builtin_amdgcn_wmma_f32_16x16x32_f16(false, a.v, false, b0,
                                                        (short)0, c0, false, false);
            c1 = __builtin_amdgcn_wmma_f32_16x16x32_f16(false, a.v, false, b1,
                                                        (short)0, c1, false, false);
        }
        __syncthreads();   // all waves done reading buf before it is re-filled
        buf ^= 1;
    }

    // epilogue: C layout lane0-15 -> M=r, lane16-31 -> M=r+8, N=lane%16
    const int mBase = rowBase + wm + (lane >> 4) * 8;
    const int n0    = colBase + wn + (lane & 15);
#pragma unroll
    for (int r = 0; r < 8; ++r) {
        const int gm = mBase + r;
        if (gm >= M) continue;
#pragma unroll
        for (int half = 0; half < 2; ++half) {
            const int gn = n0 + half * 16;
            if (gn >= N) continue;
            float v = half ? c1[r] : c0[r];
            if (bias) v += bias[gn];
            if (actElu) v = elu1(v);
            const size_t idx = (size_t)gm * (size_t)ldc + gn;
            if (res1) v += res1[idx];
            if (res2) v += res2[idx];
            C[idx] = v;
        }
    }
}

// f32 -> f16 with zero padding: dst[r*dcols+c] = (r<srows && c<scols) ? src : 0
__global__ void k_cvt16(const float* __restrict__ src, int srows, int scols,
                        _Float16* __restrict__ dst, int dcols, size_t total)
{
    const size_t i = (size_t)blockIdx.x * blockDim.x + threadIdx.x;
    if (i >= total) return;
    const int r = (int)(i / (size_t)dcols);
    const int c = (int)(i - (size_t)r * dcols);
    const float v = (r < srows && c < scols) ? src[(size_t)r * scols + c] : 0.f;
    dst[i] = (_Float16)v;
}

// ---------------------------------------------------------------------------
// LayerNorm over F=512 features: one wave32 per row, butterfly reduction.
// ---------------------------------------------------------------------------
__global__ __launch_bounds__(256) void k_layernorm(
    const float* __restrict__ x, const float* __restrict__ sc,
    const float* __restrict__ bi, float* __restrict__ y, int Nrows, int F)
{
    const int row  = blockIdx.x * 8 + (threadIdx.x >> 5);
    const int lane = threadIdx.x & 31;
    if (row >= Nrows) return;
    const float* xr = x + (size_t)row * F;
    float s = 0.f, s2 = 0.f;
    for (int j = lane; j < F; j += 32) {
        const float v = xr[j];
        s += v; s2 += v * v;
    }
#pragma unroll
    for (int o = 16; o > 0; o >>= 1) {
        s  += __shfl_xor(s, o, 32);
        s2 += __shfl_xor(s2, o, 32);
    }
    const float mean = s / (float)F;
    const float var  = s2 / (float)F - mean * mean;
    const float inv  = rsqrtf(var + 1e-5f);
    float* yr = y + (size_t)row * F;
    for (int j = lane; j < F; j += 32)
        yr[j] = (xr[j] - mean) * inv * sc[j] + bi[j];
}

__global__ void k_zero(float* __restrict__ p, size_t n) {
    const size_t i = (size_t)blockIdx.x * blockDim.x + threadIdx.x;
    if (i < n) p[i] = 0.f;
}

__global__ void k_elu(float* __restrict__ p, size_t n) {
    const size_t i = (size_t)blockIdx.x * blockDim.x + threadIdx.x;
    if (i < n) p[i] = elu1(p[i]);
}

__global__ void k_deg(const int* __restrict__ dst, float* __restrict__ deg, int E) {
    const int e = blockIdx.x * blockDim.x + threadIdx.x;
    if (e < E) atomicAdd(&deg[dst[e]], 1.f);
}

// summ[dst] += h[src]  (128 threads/edge, float4 each)
__global__ void k_scatter(const float* __restrict__ h, const int* __restrict__ src,
                          const int* __restrict__ dst, float* __restrict__ summ, int E)
{
    const size_t t = (size_t)blockIdx.x * blockDim.x + threadIdx.x;
    if (t >= (size_t)E * 128) return;
    const int e = (int)(t >> 7);
    const int q = ((int)t & 127) * 4;
    const int s = src[e], d = dst[e];
    const float4 v = *(const float4*)(h + (size_t)s * HF + q);
    float* o = summ + (size_t)d * HF + q;
    atomicAdd(o + 0, v.x);
    atomicAdd(o + 1, v.y);
    atomicAdd(o + 2, v.z);
    atomicAdd(o + 3, v.w);
}

__global__ void k_divdeg(float* __restrict__ agg, const float* __restrict__ deg, size_t n) {
    const size_t i = (size_t)blockIdx.x * blockDim.x + threadIdx.x;
    if (i < n) agg[i] /= fmaxf(deg[i >> 9], 1.f);   // i>>9 == i / HF
}

__global__ void k_attn_score(const float* __restrict__ feat, const int* __restrict__ src,
                             const int* __restrict__ dst, float* __restrict__ esc,
                             unsigned* __restrict__ mkey, int E)
{
    const int t = blockIdx.x * blockDim.x + threadIdx.x;
    if (t >= E * NHEAD) return;
    const int e = t / NHEAD, h = t - e * NHEAD;
    const int s = src[e], d = dst[e];
    const float* fs = feat + (size_t)s * FEAT + h * DPH;
    const float* fd = feat + (size_t)d * FEAT + h * DPH;
    float acc = 0.f;
#pragma unroll
    for (int j = 0; j < DPH; j += 4) {
        const float4 a = *(const float4*)(fs + j);
        const float4 b = *(const float4*)(fd + j);
        acc += a.x * b.x + a.y * b.y + a.z * b.z + a.w * b.w;
    }
    acc *= rsqrtf((float)DPH);
    esc[t] = acc;
    atomicMax(&mkey[(size_t)d * NHEAD + h], fkey(acc));
}

__global__ void k_attn_exp(float* __restrict__ esc, const unsigned* __restrict__ mkey,
                           const int* __restrict__ dst, float* __restrict__ z, int E)
{
    const int t = blockIdx.x * blockDim.x + threadIdx.x;
    if (t >= E * NHEAD) return;
    const int e = t / NHEAD, h = t - e * NHEAD;
    const int d = dst[e];
    const float m = fkeyinv(mkey[(size_t)d * NHEAD + h]);
    const float a = __expf(esc[t] - m);
    esc[t] = a;
    atomicAdd(&z[(size_t)d * NHEAD + h], a);
}

__global__ void k_attn_scatter(const float* __restrict__ esc, const float* __restrict__ z,
                               const float* __restrict__ feat, const int* __restrict__ src,
                               const int* __restrict__ dst, float* __restrict__ out, int E)
{
    const int t = blockIdx.x * blockDim.x + threadIdx.x;
    if (t >= E * NHEAD) return;
    const int e = t / NHEAD, h = t - e * NHEAD;
    const int s = src[e], d = dst[e];
    const float wgt = esc[t] / fmaxf(z[(size_t)d * NHEAD + h], 1e-9f);
    const float* fs = feat + (size_t)s * FEAT + h * DPH;
    float* o = out + (size_t)d * FEAT + h * DPH;
#pragma unroll 4
    for (int j = 0; j < DPH; ++j) atomicAdd(o + j, wgt * fs[j]);
}

// ---------------------------------------------------------------------------
extern "C" void kernel_launch(void* const* d_in, const int* in_sizes, int n_in,
                              void* d_out, int out_size, void* d_ws, size_t ws_size,
                              hipStream_t stream)
{
    (void)n_in; (void)out_size; (void)ws_size;
    const float* x     = (const float*)d_in[0];
    const int*   src   = (const int*)d_in[1];
    const int*   dst   = (const int*)d_in[2];
    const float* Wpre  = (const float*)d_in[3];
    const float* bpre  = (const float*)d_in[4];
    const float* cln_s = (const float*)d_in[5];
    const float* cln_b = (const float*)d_in[6];
    const float* Wself = (const float*)d_in[7];
    const float* Wneigh= (const float*)d_in[8];
    const float* bconv = (const float*)d_in[9];
    const float* cin_s = (const float*)d_in[10];
    const float* cin_b = (const float*)d_in[11];
    const float* Wsi   = (const float*)d_in[12];
    const float* bsi   = (const float*)d_in[13];
    const float* aln_s = (const float*)d_in[14];
    const float* aln_b = (const float*)d_in[15];
    const float* Wattn = (const float*)d_in[16];
    const float* Whr   = (const float*)d_in[17];
    const float* bhr   = (const float*)d_in[18];
    const float* ain_s = (const float*)d_in[19];
    const float* ain_b = (const float*)d_in[20];
    const float* W1    = (const float*)d_in[21];
    const float* b1    = (const float*)d_in[22];
    const float* W2    = (const float*)d_in[23];
    const float* b2    = (const float*)d_in[24];
    const float* Wpost = (const float*)d_in[25];
    const float* bpost = (const float*)d_in[26];

    const int Nn = in_sizes[0] / FIN;   // 20000
    const int E  = in_sizes[1];         // 256000

    // ---------------- workspace carve-out ----------------
    float* p = (float*)d_ws;
    auto alloc = [&](size_t n) { float* r = p; p += (n + 63) & ~(size_t)63; return r; };
    float*    h    = alloc((size_t)Nn * HF);
    float*    hn   = alloc((size_t)Nn * HF);
    float*    h2   = alloc((size_t)Nn * HF);
    float*    bufA = alloc((size_t)Nn * FEAT);   // agg  / attn feat
    float*    bufB = alloc((size_t)Nn * FEAT);   // tself/ attn out
    float*    tbig = alloc((size_t)CHUNK * 4 * HF);
    float*    esc  = alloc((size_t)E * NHEAD);
    float*    zb   = alloc((size_t)Nn * NHEAD);
    unsigned* mkey = (unsigned*)alloc((size_t)Nn * NHEAD);
    float*    deg  = alloc((size_t)Nn);

    _Float16* q = (_Float16*)p;
    auto halloc = [&](size_t n) { _Float16* r = q; q += (n + 63) & ~(size_t)63; return r; };
    _Float16* wpre16  = halloc((size_t)96 * HF);
    _Float16* wself16 = halloc((size_t)3 * HF * HF);
    _Float16* wneigh16= halloc((size_t)3 * HF * HF);
    _Float16* wsi16   = halloc((size_t)3 * HF * HF);
    _Float16* wattn16 = halloc((size_t)3 * HF * 576);
    _Float16* whr16   = halloc((size_t)3 * 544 * HF);
    _Float16* w116    = halloc((size_t)3 * HF * 4 * HF);
    _Float16* w216    = halloc((size_t)3 * 4 * HF * HF);
    _Float16* wpost16 = halloc((size_t)HF * HF);
    _Float16* x16     = halloc((size_t)MP * 96);
    _Float16* act16a  = halloc((size_t)MP * HF);
    _Float16* act16b  = halloc((size_t)MP * 544);
    _Float16* tbig16  = halloc((size_t)CHUNK * 4 * HF);

    auto cvt = [&](const float* s, int sr, int sc, _Float16* d, int dc, int dr) {
        const size_t tot = (size_t)dr * dc;
        k_cvt16<<<(unsigned)((tot + 255) / 256), 256, 0, stream>>>(s, sr, sc, d, dc, tot);
    };
    auto gemm = [&](const _Float16* A, int lda, const _Float16* B, int ldb,
                    float* C, int ldc, const float* bias, const float* r1,
                    const float* r2, int M, int Mpad, int Nc, int Kp, int act) {
        dim3 g((unsigned)((Nc + 63) / 64), (unsigned)(Mpad / 64));
        k_gemm16<<<g, 256, 0, stream>>>(A, lda, B, ldb, C, ldc, bias, r1, r2, M, Nc, Kp, act);
    };
    auto zero = [&](float* pz, size_t n) {
        k_zero<<<(unsigned)((n + 255) / 256), 256, 0, stream>>>(pz, n);
    };
    auto lnorm = [&](const float* xi, const float* s, const float* b, float* y) {
        k_layernorm<<<(Nn + 7) / 8, 256, 0, stream>>>(xi, s, b, y, Nn, HF);
    };

    // ---------------- convert + pad all weights (f32 -> f16) ----------------
    cvt(Wpre, FIN, HF, wpre16, HF, 96);
    for (int i = 0; i < 3; ++i) {
        cvt(Wself + (size_t)i * HF * HF,  HF, HF, wself16 + (size_t)i * HF * HF,  HF, HF);
        cvt(Wneigh+ (size_t)i * HF * HF,  HF, HF, wneigh16+ (size_t)i * HF * HF,  HF, HF);
        cvt(Wsi   + (size_t)i * HF * HF,  HF, HF, wsi16   + (size_t)i * HF * HF,  HF, HF);
        cvt(Wattn + (size_t)i * HF * FEAT, HF, FEAT, wattn16 + (size_t)i * HF * 576, 576, HF);
        cvt(Whr   + (size_t)i * FEAT * HF, FEAT, HF, whr16 + (size_t)i * 544 * HF, HF, 544);
        cvt(W1    + (size_t)i * HF * 4 * HF, HF, 4 * HF, w116 + (size_t)i * HF * 4 * HF, 4 * HF, HF);
        cvt(W2    + (size_t)i * 4 * HF * HF, 4 * HF, HF, w216 + (size_t)i * 4 * HF * HF, HF, 4 * HF);
    }
    cvt(Wpost, HF, HF, wpost16, HF, HF);

    // ---- pre: h = elu(x @ Wpre + bpre) ----
    cvt(x, Nn, FIN, x16, 96, MP);
    gemm(x16, 96, wpre16, HF, h, HF, bpre, nullptr, nullptr, Nn, MP, HF, 96, 1);

    // ---- residual SAGE conv blocks ----
    for (int i = 0; i < 3; ++i) {
        lnorm(h, cln_s + (size_t)i * HF, cln_b + (size_t)i * HF, hn);
        cvt(hn, Nn, HF, act16a, HF, MP);
        zero(deg, (size_t)Nn);
        zero(bufA, (size_t)Nn * HF);
        k_deg<<<(E + 255) / 256, 256, 0, stream>>>(dst, deg, E);
        k_scatter<<<(unsigned)(((size_t)E * 128 + 255) / 256), 256, 0, stream>>>(hn, src, dst, bufA, E);
        k_divdeg<<<(unsigned)(((size_t)Nn * HF + 255) / 256), 256, 0, stream>>>(bufA, deg, (size_t)Nn * HF);
        cvt(bufA, Nn, HF, act16b, HF, MP);
        // tself = hn @ Wself
        gemm(act16a, HF, wself16 + (size_t)i * HF * HF, HF, bufB, HF,
             nullptr, nullptr, nullptr, Nn, MP, HF, HF, 0);
        // h2 = agg @ Wneigh + bconv + tself + hn
        gemm(act16b, HF, wneigh16 + (size_t)i * HF * HF, HF, h2, HF,
             bconv + (size_t)i * HF, bufB, hn, Nn, MP, HF, HF, 0);
        lnorm(h2, cin_s + (size_t)i * HF, cin_b + (size_t)i * HF, hn);
        cvt(hn, Nn, HF, act16a, HF, MP);
        // h = elu(hn @ Wsi + bsi) + hn
        gemm(act16a, HF, wsi16 + (size_t)i * HF * HF, HF, h, HF,
             bsi + (size_t)i * HF, hn, nullptr, Nn, MP, HF, HF, 1);
    }

    // ---- residual attention blocks ----
    const int tE = E * NHEAD;
    for (int i = 0; i < 3; ++i) {
        lnorm(h, aln_s + (size_t)i * HF, aln_b + (size_t)i * HF, hn);
        cvt(hn, Nn, HF, act16a, HF, MP);
        // feat = hn @ Wattn   [Nn, 520]
        gemm(act16a, HF, wattn16 + (size_t)i * HF * 576, 576, bufA, FEAT,
             nullptr, nullptr, nullptr, Nn, MP, FEAT, HF, 0);
        zero((float*)mkey, (size_t)Nn * NHEAD);   // key 0 == -inf floor
        zero(zb, (size_t)Nn * NHEAD);
        zero(bufB, (size_t)Nn * FEAT);
        k_attn_score<<<(tE + 255) / 256, 256, 0, stream>>>(bufA, src, dst, esc, mkey, E);
        k_attn_exp<<<(tE + 255) / 256, 256, 0, stream>>>(esc, mkey, dst, zb, E);
        k_attn_scatter<<<(tE + 255) / 256, 256, 0, stream>>>(esc, zb, bufA, src, dst, bufB, E);
        k_elu<<<(unsigned)(((size_t)Nn * FEAT + 255) / 256), 256, 0, stream>>>(bufB, (size_t)Nn * FEAT);
        cvt(bufB, Nn, FEAT, act16b, 544, MP);
        // h2 = elu(att) @ Whr + bhr + hn
        gemm(act16b, 544, whr16 + (size_t)i * 544 * HF, HF, h2, HF,
             bhr + (size_t)i * HF, hn, nullptr, Nn, MP, HF, 544, 0);
        lnorm(h2, ain_s + (size_t)i * HF, ain_b + (size_t)i * HF, hn);
        cvt(hn, Nn, HF, act16a, HF, MP);
        // FFN (row-chunked): h = elu(elu(hn@W1+b1)@W2+b2) + hn
        for (int c0 = 0; c0 < Nn; c0 += CHUNK) {
            const int rows = (Nn - c0 < CHUNK) ? (Nn - c0) : CHUNK;
            const int rowsPad = (rows + 63) & ~63;
            gemm(act16a + (size_t)c0 * HF, HF, w116 + (size_t)i * HF * 4 * HF, 4 * HF,
                 tbig, 4 * HF, b1 + (size_t)i * 4 * HF, nullptr, nullptr,
                 rows, rowsPad, 4 * HF, HF, 1);
            cvt(tbig, rows, 4 * HF, tbig16, 4 * HF, rowsPad);
            gemm(tbig16, 4 * HF, w216 + (size_t)i * 4 * HF * HF, HF,
                 h + (size_t)c0 * HF, HF, b2 + (size_t)i * HF,
                 hn + (size_t)c0 * HF, nullptr, rows, rowsPad, HF, 4 * HF, 1);
        }
    }

    // ---- post: out = h @ Wpost + bpost ----
    cvt(h, Nn, HF, act16a, HF, MP);
    gemm(act16a, HF, wpost16, HF, (float*)d_out, HF, bpost, nullptr, nullptr,
         Nn, MP, HF, HF, 0);
}
